// GPS_75187697484287
// MI455X (gfx1250) — matrix operand order repaired and checked by reading, not compiled, not run on hardware
//
#include <hip/hip_runtime.h>
#include <hip/hip_bf16.h>

// ---------------------------------------------------------------------------
// GraphGPS forward on gfx1250 (MI455X). All dense math runs through
// v_wmma_f32_16x16x32_bf16 (fp32 accumulate), wave32, 8-wave blocks.
// Data movement: GLOBAL_LOAD_ASYNC_TO_LDS_B128 double-buffering (ASYNCcnt),
// b128 vector loads, ds_swizzle butterflies, fp32 atomics for edge scatter.
// ---------------------------------------------------------------------------

typedef __attribute__((ext_vector_type(16))) __bf16 v16bf;
typedef __attribute__((ext_vector_type(8)))  float  v8f;

__device__ __forceinline__ v8f wmma_bf16(v16bf a, v16bf b, v8f c) {
  // 8 args: (neg_a, A, neg_b, B, c_mod, C, reuse_a, reuse_b)
  return __builtin_amdgcn_wmma_f32_16x16x32_bf16(false, a, false, b,
                                                 (short)0, c, false, false);
}

// ds_swizzle group-of-32 mode (offset[15]=0): and=0x1f, or=0, xor=XMASK.
// One LDS-permute instruction per butterfly level, no clamping cndmasks.
template <int XMASK>
__device__ __forceinline__ float swz_max16(float v) {
  int w = __builtin_amdgcn_ds_swizzle(__float_as_int(v), (XMASK << 10) | 0x1f);
  return fmaxf(v, __int_as_float(w));
}

// LDS byte offset of a __shared__ object: LDS aperture occupies addr[63:32]
// (ISA 10.2), so the low 32 bits of the flat address are the wave-relative
// LDS offset required by the async-load VDST operand.
template <typename T>
__device__ __forceinline__ uint32_t lds_off(T* p) {
  return (uint32_t)(uintptr_t)p;
}

// GVS-mode async copy: LDS[dst] <- MEM[saddr + voff], 16 bytes. ASYNCcnt++.
__device__ __forceinline__ void async_copy_b128(uint32_t dst, uint32_t voff,
                                                const void* saddr) {
  asm volatile("global_load_async_to_lds_b128 %0, %1, %2"
               :: "v"(dst), "v"(voff), "s"(saddr) : "memory");
}
__device__ __forceinline__ void wait_async0() {
  asm volatile("s_wait_asynccnt 0" ::: "memory");
}

// ---------------------------------------------------------------------------
// Generic WMMA GEMM:  OUT[N, NT*16] = act((A [+A2]) @ W + bias) [+ R]
//   A: [Nrows, Kdim] fp32 row-major.
//   W: wTrans==0 -> [Kdim, Ncols]; wTrans==1 -> [Ncols, Kdim].
//   Weights async-DMA'd (b128) into LDS raw, converted once to bf16
//   column-major [Ncols][Kpad] so each lane's B fragment is two contiguous
//   16B LDS reads. Block 256 thr (8 waves); wave w owns rows blk*128+w*16.
// ---------------------------------------------------------------------------
template <int NT>
__global__ __launch_bounds__(256)
void gemm_wmma_kernel(const float* __restrict__ A,
                      const float* __restrict__ A2,
                      const float* __restrict__ W,
                      const float* __restrict__ bias,
                      const float* __restrict__ R,
                      float* __restrict__ OUT,
                      int Nrows, int Kdim, int wTrans, int relu)
{
  constexpr int Ncols = NT * 16;
  const int Kpad = (Kdim + 31) & ~31;
  extern __shared__ char smem_raw[];
  float*  Wraw = (float*)smem_raw;                                   // [Kdim*Ncols]
  __bf16* Wl   = (__bf16*)(smem_raw + (size_t)Kdim * Ncols * 4);     // [Ncols][Kpad]

  const int tid = threadIdx.x;

  // ---- phase 1: async DMA of the raw fp32 weight block into LDS
  const int nflt = Kdim * Ncols;                 // always a multiple of 4
  for (int idx = tid * 4; idx < nflt; idx += 1024)
    async_copy_b128(lds_off(Wraw + idx), (uint32_t)idx * 4u, W);
  wait_async0();
  __syncthreads();

  // ---- phase 2: convert to bf16, transposed+padded B layout
  for (int idx = tid; idx < Ncols * Kpad; idx += 256) {
    int n = idx / Kpad, k = idx - n * Kpad;
    float v = 0.f;
    if (k < Kdim) v = wTrans ? Wraw[n * Kdim + k] : Wraw[k * Ncols + n];
    Wl[idx] = (__bf16)v;
  }
  __syncthreads();

  const int wave = tid >> 5;
  const int lane = tid & 31;
  const int half = lane >> 4;
  const int l16  = lane & 15;
  const int row0 = blockIdx.x * 128 + wave * 16;
  const int arow = row0 + l16;
  const bool kexact = (Kdim & 31) == 0;          // Kpad == Kdim -> b128 path

  v8f acc[NT];
#pragma unroll
  for (int t = 0; t < NT; ++t)
    acc[t] = v8f{0.f, 0.f, 0.f, 0.f, 0.f, 0.f, 0.f, 0.f};

  for (int k0 = 0; k0 < Kpad; k0 += 32) {
    // ---- A fragment (16x32): lane row = l16; K = k0+8*half+{0..7, 16..23}
    const int kb = k0 + half * 8;
    float va[16];
    if (kexact) {
      const float* ap = A + (size_t)arow * Kdim + kb;
      float4 a0 = *(const float4*)(ap + 0);
      float4 a1 = *(const float4*)(ap + 4);
      float4 a2 = *(const float4*)(ap + 16);
      float4 a3 = *(const float4*)(ap + 20);
      va[0]=a0.x; va[1]=a0.y; va[2]=a0.z;  va[3]=a0.w;
      va[4]=a1.x; va[5]=a1.y; va[6]=a1.z;  va[7]=a1.w;
      va[8]=a2.x; va[9]=a2.y; va[10]=a2.z; va[11]=a2.w;
      va[12]=a3.x; va[13]=a3.y; va[14]=a3.z; va[15]=a3.w;
      if (A2) {
        const float* bp2 = A2 + (size_t)arow * Kdim + kb;
        float4 c0 = *(const float4*)(bp2 + 0);
        float4 c1 = *(const float4*)(bp2 + 4);
        float4 c2 = *(const float4*)(bp2 + 16);
        float4 c3 = *(const float4*)(bp2 + 20);
        va[0]+=c0.x; va[1]+=c0.y; va[2]+=c0.z;  va[3]+=c0.w;
        va[4]+=c1.x; va[5]+=c1.y; va[6]+=c1.z;  va[7]+=c1.w;
        va[8]+=c2.x; va[9]+=c2.y; va[10]+=c2.z; va[11]+=c2.w;
        va[12]+=c3.x; va[13]+=c3.y; va[14]+=c3.z; va[15]+=c3.w;
      }
    } else {
#pragma unroll
      for (int j = 0; j < 8; ++j) {
        int k1 = kb + j, k2 = kb + 16 + j;
        float v1 = (k1 < Kdim) ? A[(size_t)arow * Kdim + k1] : 0.f;
        float v2 = (k2 < Kdim) ? A[(size_t)arow * Kdim + k2] : 0.f;
        if (A2) {
          if (k1 < Kdim) v1 += A2[(size_t)arow * Kdim + k1];
          if (k2 < Kdim) v2 += A2[(size_t)arow * Kdim + k2];
        }
        va[j] = v1; va[8 + j] = v2;
      }
    }
    v16bf afrag;
#pragma unroll
    for (int j = 0; j < 16; ++j) afrag[j] = (__bf16)va[j];

#pragma unroll
    for (int t = 0; t < NT; ++t) {
      // ---- B fragment (32x16): lane col = l16; K = k0+8*half+{0..7, 16..23}
      v16bf bfrag;
      const __bf16* bp = &Wl[(t * 16 + l16) * Kpad + k0 + half * 8];
#pragma unroll
      for (int j = 0; j < 16; ++j) bfrag[j] = bp[(j & 7) + (j >> 3) * 16];
      acc[t] = wmma_bf16(afrag, bfrag, acc[t]);
    }
  }

  // ---- epilogue: C/D layout: lane col = l16, rows r + 8*half
#pragma unroll
  for (int t = 0; t < NT; ++t) {
    const int col = t * 16 + l16;
    const float bv = bias ? bias[col] : 0.f;
#pragma unroll
    for (int r = 0; r < 8; ++r) {
      const int row = row0 + r + half * 8;
      float v = acc[t][r] + bv;
      if (relu) v = fmaxf(v, 0.f);
      if (R) v += R[(size_t)row * Ncols + col];
      OUT[(size_t)row * Ncols + col] = v;
    }
  }
}

// ---------------------------------------------------------------------------
// Flash attention: H=4, D=16. grid = (N/128, H), block = 256 (8 waves);
// each wave owns 16 query rows. 32-key K/V tiles double-buffered in LDS via
// async b128 copies (one per thread per tile) overlapping the WMMAs.
// Row-max: ds_swizzle xor-butterfly. Row-sum: P x ones WMMA (each lane gets
// its 8 rows' sums directly in the C/D layout -> zero shuffles).
// ---------------------------------------------------------------------------
__global__ __launch_bounds__(256)
void attn_flash_kernel(const float* __restrict__ qkv,  // [N, 192]
                       float* __restrict__ o,          // [N, 64]
                       int Nn)
{
  const int h    = blockIdx.y;
  const int tid  = threadIdx.x;
  const int wave = tid >> 5;
  const int lane = tid & 31;
  const int half = lane >> 4;
  const int l16  = lane & 15;
  const int q0   = blockIdx.x * 128 + wave * 16;
  const float scale = 0.25f;                     // 1/sqrt(D)

  __shared__ __align__(16) float Kraw[2][32][16];   // [buf][key][d]
  __shared__ __align__(16) float Vraw[2][32][16];   // [buf][key][d]
  __shared__ __bf16 Pl[8][16][32];                  // per-wave P scratch

  // per-thread async-staging role: 32 keys x (K chunk + V chunk) x 4 b128
  const int key  = tid >> 3;
  const int part = tid & 7;
  const int isV  = part >> 2;
  const int d4   = (part & 3) * 4;
  float* dstp = isV ? &Vraw[0][key][d4] : &Kraw[0][key][d4];
  const uint32_t dst0   = lds_off(dstp);
  const uint32_t dstStr = lds_off(&Kraw[1][0][0]) - lds_off(&Kraw[0][0][0]);

  auto stage_tile = [&](int j0, int buf) {
    uint32_t src = (uint32_t)(((j0 + key) * 192 + 64 + isV * 64 + h * 16 + d4) * 4);
    async_copy_b128(dst0 + buf * dstStr, src, qkv);
  };

  // Q fragment: row q0+l16; K = 16*half + j; lane half 1 (K=16..31) is zero pad
  v16bf qf;
  {
    const float* qp = &qkv[(size_t)(q0 + l16) * 192 + h * 16];
#pragma unroll
    for (int j = 0; j < 16; ++j)
      qf[j] = (__bf16)(half == 0 ? qp[j] * scale : 0.f);
  }
  v16bf onef;                                     // all-ones B for row sums
#pragma unroll
  for (int j = 0; j < 16; ++j) onef[j] = (__bf16)1.0f;
  const v8f zero8 = v8f{0.f, 0.f, 0.f, 0.f, 0.f, 0.f, 0.f, 0.f};

  float mrow[8], lrow[8];
  v8f oacc = zero8;
#pragma unroll
  for (int r = 0; r < 8; ++r) { mrow[r] = -1e30f; lrow[r] = 0.f; }

  stage_tile(0, 0);
  wait_async0();
  __syncthreads();

  const int ntiles = Nn >> 5;
  for (int it = 0; it < ntiles; ++it) {
    const int cur = it & 1;
    if (it + 1 < ntiles) stage_tile((it + 1) << 5, cur ^ 1);  // overlap DMA

    // ---- two QK^T WMMAs (key groups of 16)
    v8f s[2];
#pragma unroll
    for (int g = 0; g < 2; ++g) {
      v16bf bf;
      if (half == 0) {                           // K = 0..15 (valid D)
        const float* kp = &Kraw[cur][g * 16 + l16][0];
#pragma unroll
        for (int j = 0; j < 16; ++j) bf[j] = (__bf16)kp[j];
      } else {                                   // K = 16..31 (pad)
#pragma unroll
        for (int j = 0; j < 16; ++j) bf[j] = (__bf16)0.f;
      }
      s[g] = wmma_bf16(qf, bf, zero8);
    }

    // ---- online softmax: row-max by ds_swizzle butterfly (16-lane groups)
    float corr[8];
#pragma unroll
    for (int r = 0; r < 8; ++r) {
      float mx = fmaxf(s[0][r], s[1][r]);
      mx = swz_max16<1>(mx); mx = swz_max16<2>(mx);
      mx = swz_max16<4>(mx); mx = swz_max16<8>(mx);
      float nm = fmaxf(mrow[r], mx);
      corr[r]  = __expf(mrow[r] - nm);
      float p0 = __expf(s[0][r] - nm);
      float p1 = __expf(s[1][r] - nm);
      mrow[r]  = nm;
      oacc[r] *= corr[r];
      Pl[wave][r + half * 8][l16]      = (__bf16)p0;   // C-layout -> LDS
      Pl[wave][r + half * 8][16 + l16] = (__bf16)p1;
    }

    // ---- P (A-layout 16x32) from LDS
    v16bf pf;
#pragma unroll
    for (int j = 0; j < 8; ++j) {
      pf[j]     = Pl[wave][l16][half * 8 + j];
      pf[j + 8] = Pl[wave][l16][16 + half * 8 + j];
    }

    // ---- row sums via P x ones WMMA (no cross-lane shuffles needed)
    v8f rsum = wmma_bf16(pf, onef, zero8);
#pragma unroll
    for (int r = 0; r < 8; ++r) lrow[r] = lrow[r] * corr[r] + rsum[r];

    // ---- V (B-layout 32x16, transposed on read) and O accumulate
    v16bf vf;
#pragma unroll
    for (int j = 0; j < 16; ++j) vf[j] = (__bf16)Vraw[cur][half * 8 + (j & 7) + (j >> 3) * 16][l16];
    oacc = wmma_bf16(pf, vf, oacc);

    wait_async0();          // next tile landed in the alternate buffer
    __syncthreads();        // everyone done reading buffer `cur`
  }

#pragma unroll
  for (int r = 0; r < 8; ++r) {
    const int row = q0 + r + half * 8;
    o[(size_t)row * 64 + h * 16 + l16] = oacc[r] / lrow[r];
  }
}

// ---------------------------------------------------------------------------
// Edge scatter-sum:  agg[dst] += x[src]   (fp32 atomics, 4 channels/thread)
// ---------------------------------------------------------------------------
__global__ void zero4_kernel(float4* __restrict__ p, int n4) {
  int i = blockIdx.x * blockDim.x + threadIdx.x;
  if (i < n4) p[i] = float4{0.f, 0.f, 0.f, 0.f};
}

__global__ void scatter_kernel(const float* __restrict__ x,
                               const long long* __restrict__ src,
                               const long long* __restrict__ dst,
                               float* __restrict__ agg, int E)
{
  int t = blockIdx.x * blockDim.x + threadIdx.x;
  int e = t >> 4;
  if (e >= E) return;
  int c4 = (t & 15) * 4;
  int e2 = e + 4096;                    // prefetch a future gather row
  if (e2 < E) __builtin_prefetch(&x[(size_t)(int)src[e2] * 64], 0, 1);
  int s = (int)src[e], d = (int)dst[e];
  const float4 v = *(const float4*)&x[(size_t)s * 64 + c4];
  atomicAdd(&agg[(size_t)d * 64 + c4 + 0], v.x);
  atomicAdd(&agg[(size_t)d * 64 + c4 + 1], v.y);
  atomicAdd(&agg[(size_t)d * 64 + c4 + 2], v.z);
  atomicAdd(&agg[(size_t)d * 64 + c4 + 3], v.w);
}

// ---------------------------------------------------------------------------
// Per-column stats (mean, E[x^2]) over Nrows — one block per 4 columns,
// float4 (b128) loads.
// ---------------------------------------------------------------------------
__global__ __launch_bounds__(256)
void colstats4_kernel(const float* __restrict__ X, int Nrows, int Ncols,
                      float* __restrict__ mean, float* __restrict__ ex2)
{
  __shared__ float sh0[256][4], sh1[256][4];
  const int c4 = blockIdx.x * 4;
  const int tid = threadIdx.x;
  float s0 = 0.f, s1 = 0.f, s2 = 0.f, s3 = 0.f;
  float q0 = 0.f, q1 = 0.f, q2 = 0.f, q3 = 0.f;
  for (int r = tid; r < Nrows; r += 256) {
    float4 v = *(const float4*)&X[(size_t)r * Ncols + c4];
    s0 += v.x; q0 += v.x * v.x;
    s1 += v.y; q1 += v.y * v.y;
    s2 += v.z; q2 += v.z * v.z;
    s3 += v.w; q3 += v.w * v.w;
  }
  sh0[tid][0]=s0; sh0[tid][1]=s1; sh0[tid][2]=s2; sh0[tid][3]=s3;
  sh1[tid][0]=q0; sh1[tid][1]=q1; sh1[tid][2]=q2; sh1[tid][3]=q3;
  __syncthreads();
  for (int d = 128; d > 0; d >>= 1) {
    if (tid < d) {
#pragma unroll
      for (int j = 0; j < 4; ++j) {
        sh0[tid][j] += sh0[tid + d][j];
        sh1[tid][j] += sh1[tid + d][j];
      }
    }
    __syncthreads();
  }
  if (tid < 4) {
    mean[c4 + tid] = sh0[0][tid] / Nrows;
    ex2[c4 + tid]  = sh1[0][tid] / Nrows;
  }
}

// BatchNorm apply (float4): OUT = g*(X-m)*rsqrt(E[x^2]-m^2+eps)+b  [+ add]
__global__ void bn_apply4_kernel(const float4* __restrict__ X,
                                 const float* __restrict__ g,
                                 const float* __restrict__ b,
                                 const float* __restrict__ mean,
                                 const float* __restrict__ ex2,
                                 const float4* __restrict__ add,
                                 float4* __restrict__ OUT, int n4, int Ncols)
{
  int i = blockIdx.x * blockDim.x + threadIdx.x;
  if (i >= n4) return;
  int c4 = (i * 4) & (Ncols - 1);
  float4 xv = X[i];
  float4 gv = *(const float4*)&g[c4];
  float4 bv = *(const float4*)&b[c4];
  float4 mv = *(const float4*)&mean[c4];
  float4 ev = *(const float4*)&ex2[c4];
  float4 ov;
  ov.x = gv.x * (xv.x - mv.x) * rsqrtf(ev.x - mv.x * mv.x + 1e-5f) + bv.x;
  ov.y = gv.y * (xv.y - mv.y) * rsqrtf(ev.y - mv.y * mv.y + 1e-5f) + bv.y;
  ov.z = gv.z * (xv.z - mv.z) * rsqrtf(ev.z - mv.z * mv.z + 1e-5f) + bv.z;
  ov.w = gv.w * (xv.w - mv.w) * rsqrtf(ev.w - mv.w * mv.w + 1e-5f) + bv.w;
  if (add) {
    float4 av = add[i];
    ov.x += av.x; ov.y += av.y; ov.z += av.z; ov.w += av.w;
  }
  OUT[i] = ov;
}

// GraphNorm + ReLU (float4): cen = X - ms*m; var = E[x^2]-2*ms*m^2+ms^2*m^2
__global__ void gn_apply4_kernel(const float4* __restrict__ X,
                                 const float* __restrict__ g,
                                 const float* __restrict__ b,
                                 const float* __restrict__ ms,
                                 const float* __restrict__ mean,
                                 const float* __restrict__ ex2,
                                 float4* __restrict__ OUT, int n4, int Ncols)
{
  int i = blockIdx.x * blockDim.x + threadIdx.x;
  if (i >= n4) return;
  int c4 = (i * 4) & (Ncols - 1);
  float4 xv = X[i];
  float4 gv = *(const float4*)&g[c4];
  float4 bv = *(const float4*)&b[c4];
  float4 sv = *(const float4*)&ms[c4];
  float4 mv = *(const float4*)&mean[c4];
  float4 ev = *(const float4*)&ex2[c4];
  float4 ov;
#define GN1(f) do { \
    float var = ev.f - 2.f * sv.f * mv.f * mv.f + sv.f * sv.f * mv.f * mv.f; \
    ov.f = fmaxf(gv.f * (xv.f - sv.f * mv.f) * rsqrtf(var + 1e-5f) + bv.f, 0.f); \
  } while (0)
  GN1(x); GN1(y); GN1(z); GN1(w);
#undef GN1
  OUT[i] = ov;
}

// ---------------------------------------------------------------------------
// MLP head (64 -> 32 -> 16 -> 1), one row per thread (tiny).
// ---------------------------------------------------------------------------
__global__ __launch_bounds__(256)
void head_kernel(const float* __restrict__ x,
                 const float* __restrict__ w1, const float* __restrict__ b1,
                 const float* __restrict__ w2, const float* __restrict__ b2,
                 const float* __restrict__ w3, const float* __restrict__ b3,
                 float* __restrict__ out, int Nn)
{
  int r = blockIdx.x * blockDim.x + threadIdx.x;
  if (r >= Nn) return;
  float xr[64];
#pragma unroll
  for (int c = 0; c < 64; ++c) xr[c] = x[(size_t)r * 64 + c];
  float h1[32];
  for (int j = 0; j < 32; ++j) {
    float s = b1[j];
    for (int c = 0; c < 64; ++c) s += xr[c] * w1[c * 32 + j];
    h1[j] = fmaxf(s, 0.f);
  }
  float h2[16];
  for (int j = 0; j < 16; ++j) {
    float s = b2[j];
    for (int c = 0; c < 32; ++c) s += h1[c] * w2[c * 16 + j];
    h2[j] = fmaxf(s, 0.f);
  }
  float s = b3[0];
  for (int c = 0; c < 16; ++c) s += h2[c] * w3[c];
  out[r] = s;
}

// ---------------------------------------------------------------------------
// Host orchestration
// ---------------------------------------------------------------------------
extern "C" void kernel_launch(void* const* d_in, const int* in_sizes, int n_in,
                              void* d_out, int out_size, void* d_ws, size_t ws_size,
                              hipStream_t stream)
{
  (void)in_sizes; (void)n_in; (void)out_size; (void)ws_size;
  const int N = 4096, E = 262144, IN_CH = 165, C = 64, L = 4;

  const float*     x_in   = (const float*)d_in[0];
  const long long* ei     = (const long long*)d_in[1];
  const long long* e_src  = ei;
  const long long* e_dst  = ei + E;
  const float* init_w     = (const float*)d_in[2];
  const float* init_b     = (const float*)d_in[3];
  const float* gin_w1     = (const float*)d_in[4];
  const float* gin_b1     = (const float*)d_in[5];
  const float* gin_w2     = (const float*)d_in[6];
  const float* gin_b2     = (const float*)d_in[7];
  const float* attn_in_w  = (const float*)d_in[8];
  const float* attn_in_b  = (const float*)d_in[9];
  const float* attn_out_w = (const float*)d_in[10];
  const float* attn_out_b = (const float*)d_in[11];
  const float* norm1_g    = (const float*)d_in[12];
  const float* norm1_b    = (const float*)d_in[13];
  const float* norm2_g    = (const float*)d_in[14];
  const float* norm2_b    = (const float*)d_in[15];
  const float* norm3_g    = (const float*)d_in[16];
  const float* norm3_b    = (const float*)d_in[17];
  const float* mlp_w1     = (const float*)d_in[18];
  const float* mlp_b1     = (const float*)d_in[19];
  const float* mlp_w2     = (const float*)d_in[20];
  const float* mlp_b2     = (const float*)d_in[21];
  const float* gn_w       = (const float*)d_in[22];
  const float* gn_b       = (const float*)d_in[23];
  const float* gn_ms      = (const float*)d_in[24];
  const float* head_w1    = (const float*)d_in[25];
  const float* head_b1    = (const float*)d_in[26];
  const float* head_w2    = (const float*)d_in[27];
  const float* head_b2    = (const float*)d_in[28];
  const float* head_w3    = (const float*)d_in[29];
  const float* head_b3    = (const float*)d_in[30];

  float* ws = (float*)d_ws;
  size_t off = 0;
  float* x    = ws + off; off += (size_t)N * C;
  float* agg  = ws + off; off += (size_t)N * C;
  float* t1   = ws + off; off += (size_t)N * C;
  float* t2   = ws + off; off += (size_t)N * C;
  float* hbn  = ws + off; off += (size_t)N * C;
  float* qkv  = ws + off; off += (size_t)N * 3 * C;
  float* oatt = ws + off; off += (size_t)N * C;
  float* tb   = ws + off; off += (size_t)N * C;
  float* comb = ws + off; off += (size_t)N * C;
  float* mid  = ws + off; off += (size_t)N * 2 * C;
  float* pre3 = ws + off; off += (size_t)N * C;
  float* mean = ws + off; off += 64;
  float* ex2  = ws + off; off += 64;

  const dim3 blk(256);
  const int gGemm = N / 128;        // 32 blocks, 8 waves each -> 128 rows/block
  const int n4    = (N * C) / 4;
  const int g4    = n4 / 256;       // float4 elementwise grid

  // dynamic LDS bytes for gemm: raw fp32 W + converted bf16 [Ncols][Kpad]
  auto glds = [](int Kdim, int Ncols) -> size_t {
    int Kpad = (Kdim + 31) & ~31;
    return (size_t)Kdim * Ncols * 4 + (size_t)Ncols * Kpad * 2;
  };

  // ---- initial projection: x = x_in @ init_w + init_b  (K=165, pad->192)
  gemm_wmma_kernel<4><<<gGemm, blk, glds(IN_CH, C), stream>>>(
      x_in, nullptr, init_w, init_b, nullptr, x, N, IN_CH, 0, 0);

  for (int i = 0; i < L; ++i) {
    // ---- GIN branch -------------------------------------------------------
    zero4_kernel<<<g4, blk, 0, stream>>>((float4*)agg, n4);
    scatter_kernel<<<(E * 16) / 256, blk, 0, stream>>>(x, e_src, e_dst, agg, E);
    gemm_wmma_kernel<4><<<gGemm, blk, glds(C, C), stream>>>(
        x, agg, gin_w1 + (size_t)i * C * C, gin_b1 + (size_t)i * C,
        nullptr, t1, N, C, 0, 1);
    gemm_wmma_kernel<4><<<gGemm, blk, glds(C, C), stream>>>(
        t1, nullptr, gin_w2 + (size_t)i * C * C, gin_b2 + (size_t)i * C,
        x, t2, N, C, 0, 0);
    colstats4_kernel<<<C / 4, blk, 0, stream>>>(t2, N, C, mean, ex2);
    bn_apply4_kernel<<<g4, blk, 0, stream>>>(
        (const float4*)t2, norm1_g + (size_t)i * C, norm1_b + (size_t)i * C,
        mean, ex2, nullptr, (float4*)hbn, n4, C);

    // ---- attention branch -------------------------------------------------
    gemm_wmma_kernel<12><<<gGemm, blk, glds(C, 3 * C), stream>>>(
        x, nullptr, attn_in_w + (size_t)i * 3 * C * C, attn_in_b + (size_t)i * 3 * C,
        nullptr, qkv, N, C, 1, 0);
    attn_flash_kernel<<<dim3(N / 128, 4), blk, 0, stream>>>(qkv, oatt, N);
    gemm_wmma_kernel<4><<<gGemm, blk, glds(C, C), stream>>>(
        oatt, nullptr, attn_out_w + (size_t)i * C * C, attn_out_b + (size_t)i * C,
        x, tb, N, C, 1, 0);
    colstats4_kernel<<<C / 4, blk, 0, stream>>>(tb, N, C, mean, ex2);
    bn_apply4_kernel<<<g4, blk, 0, stream>>>(
        (const float4*)tb, norm2_g + (size_t)i * C, norm2_b + (size_t)i * C,
        mean, ex2, (const float4*)hbn, (float4*)comb, n4, C);  // comb = bn(tb)+hbn

    // ---- feedforward ------------------------------------------------------
    gemm_wmma_kernel<8><<<gGemm, blk, glds(C, 2 * C), stream>>>(
        comb, nullptr, mlp_w1 + (size_t)i * C * 2 * C, mlp_b1 + (size_t)i * 2 * C,
        nullptr, mid, N, C, 0, 1);
    gemm_wmma_kernel<4><<<gGemm, blk, glds(2 * C, C), stream>>>(
        mid, nullptr, mlp_w2 + (size_t)i * 2 * C * C, mlp_b2 + (size_t)i * C,
        comb, pre3, N, 2 * C, 0, 0);
    colstats4_kernel<<<C / 4, blk, 0, stream>>>(pre3, N, C, mean, ex2);
    bn_apply4_kernel<<<g4, blk, 0, stream>>>(
        (const float4*)pre3, norm3_g + (size_t)i * C, norm3_b + (size_t)i * C,
        mean, ex2, nullptr, (float4*)t2, n4, C);

    // ---- GraphNorm + ReLU -> next x --------------------------------------
    colstats4_kernel<<<C / 4, blk, 0, stream>>>(t2, N, C, mean, ex2);
    gn_apply4_kernel<<<g4, blk, 0, stream>>>(
        (const float4*)t2, gn_w + (size_t)i * C, gn_b + (size_t)i * C,
        gn_ms + (size_t)i * C, mean, ex2, (float4*)x, n4, C);
  }

  // ---- head ---------------------------------------------------------------
  head_kernel<<<N / 256, blk, 0, stream>>>(
      x, head_w1, head_b1, head_w2, head_b2, head_w3, head_b3,
      (float*)d_out, N);
}
